// CLIPAttention_88965952569974
// MI455X (gfx1250) — compile-verified, matrix operand order checked by reading, not run on hardware
//
#include <hip/hip_runtime.h>

// CLIP attention for MI455X (gfx1250, wave32, WMMA f32_16x16x32_f16)
// B=4, S=2048, E=1024, H=16, D=64

#define B_ 4
#define S_ 2048
#define E_ 1024
#define H_ 16
#define D_ 64
#define SCALE_ 0.125f

typedef _Float16 h16;
typedef __attribute__((ext_vector_type(4)))  float    v4f;
typedef __attribute__((ext_vector_type(8)))  float    v8f;
typedef __attribute__((ext_vector_type(8)))  _Float16 v8h;
typedef __attribute__((ext_vector_type(16))) _Float16 v16h;

// ---------------------------------------------------------------------------
// WMMA helpers
// ---------------------------------------------------------------------------
__device__ __forceinline__ v8f wmma_f16(v16h a, v16h b, v8f c) {
  // 8 args: (neg_a, A, neg_b, B, c_mod, C, reuse_a, reuse_b)
  return __builtin_amdgcn_wmma_f32_16x16x32_f16(
      false, a, false, b, (short)0, c, false, false);
}

// Load one A/B fragment in the ISA 16-bit 16x32 layout:
// per lane: 8 contiguous halves at p, 8 more at p+16
__device__ __forceinline__ v16h ldfrag(const h16* p) {
  v8h lo = *(const v8h*)p;
  v8h hi = *(const v8h*)(p + 16);
  return __builtin_shufflevector(lo, hi,
      0, 1, 2, 3, 4, 5, 6, 7, 8, 9, 10, 11, 12, 13, 14, 15);
}

// GEMM tile config
#define BM 128
#define BN 128
#define BK 32
#define NKT (E_ / BK)
#define LDT 48   // LDS row stride in halves (96 B, keeps b128 alignment)

// ---------------------------------------------------------------------------
// Kernel 1: fused Q/K/V projection (double-buffered LDS).
//   out = (X @ W^T + bias) * scale, cast to f16, written as [B,H,S,D]
// ---------------------------------------------------------------------------
__global__ void __launch_bounds__(256)
qkv_kernel(const float* __restrict__ X,
           const float* __restrict__ Wq, const float* __restrict__ Wk,
           const float* __restrict__ Wv,
           const float* __restrict__ bq, const float* __restrict__ bk,
           const float* __restrict__ bv,
           h16* __restrict__ Qh, h16* __restrict__ Kh, h16* __restrict__ Vh)
{
  __shared__ h16 At[2][BM * LDT];
  __shared__ h16 Bt[2][BN * LDT];

  const float* W; const float* bias; h16* out; float scale;
  if (blockIdx.z == 0)      { W = Wq; bias = bq; out = Qh; scale = SCALE_; }
  else if (blockIdx.z == 1) { W = Wk; bias = bk; out = Kh; scale = 1.0f; }
  else                      { W = Wv; bias = bv; out = Vh; scale = 1.0f; }

  const int tid  = threadIdx.x;
  const int lane = tid & 31;
  const int wave = tid >> 5;
  const int wm   = wave >> 2;   // 0..1 : 64-row slab
  const int wn   = wave & 3;    // 0..3 : 32-col slab
  const int sub  = lane >> 4;   // half-wave
  const int l16  = lane & 15;

  const int m0 = blockIdx.x * BM;
  const int n0 = blockIdx.y * BN;

  float bv2[2];
#pragma unroll
  for (int nf = 0; nf < 2; ++nf)
    bv2[nf] = bias[n0 + wn * 32 + nf * 16 + l16];

  v8f acc[4][2] = {};

  const int lr = tid >> 3;        // 0..31
  const int lc = (tid & 7) * 4;   // 0..28 (floats)

  // fp32 tile -> f16 LDS (VALU cvt path), uniform control flow
  auto load_tile = [&](int bsel, int k0) {
#pragma unroll
    for (int it = 0; it < 4; ++it) {
      int rr = lr + it * 32;
      v4f xa = *(const v4f*)(X + (size_t)(m0 + rr) * E_ + k0 + lc);
      v4f wa = *(const v4f*)(W + (size_t)(n0 + rr) * E_ + k0 + lc);
      h16* pa = At[bsel] + rr * LDT + lc;
      h16* pb = Bt[bsel] + rr * LDT + lc;
#pragma unroll
      for (int i = 0; i < 4; ++i) { pa[i] = (h16)xa[i]; pb[i] = (h16)wa[i]; }
    }
  };

  load_tile(0, 0);
  __syncthreads();

  int bsel = 0;
  for (int kt = 0; kt < NKT; ++kt) {
    if (kt + 1 < NKT) load_tile(bsel ^ 1, (kt + 1) * BK);

    v16h bf[2];
#pragma unroll
    for (int nf = 0; nf < 2; ++nf)
      bf[nf] = ldfrag(Bt[bsel] + (wn * 32 + nf * 16 + l16) * LDT + sub * 8);
#pragma unroll
    for (int mf = 0; mf < 4; ++mf) {
      v16h af = ldfrag(At[bsel] + (wm * 64 + mf * 16 + l16) * LDT + sub * 8);
#pragma unroll
      for (int nf = 0; nf < 2; ++nf)
        acc[mf][nf] = wmma_f16(af, bf[nf], acc[mf][nf]);
    }
    __syncthreads();
    bsel ^= 1;
  }

  // epilogue: (acc + bias) * scale -> f16, [B,H,S,D]
#pragma unroll
  for (int mf = 0; mf < 4; ++mf) {
#pragma unroll
    for (int nf = 0; nf < 2; ++nf) {
#pragma unroll
      for (int r = 0; r < 8; ++r) {
        int m = m0 + wm * 64 + mf * 16 + r + sub * 8;   // b*S + s
        int n = n0 + wn * 32 + nf * 16 + l16;           // h*D + d
        float v = (acc[mf][nf][r] + bv2[nf]) * scale;
        size_t idx = (((size_t)(m >> 11) * H_ + (n >> 6)) * S_ +
                      (m & (S_ - 1))) * D_ + (n & (D_ - 1));
        out[idx] = (h16)v;
      }
    }
  }
}

// ---------------------------------------------------------------------------
// Kernel 2: streaming (flash) attention per (b,h), double-buffered V tile.
//   8 waves/block, each wave owns 16 query rows (block covers 128 rows).
// ---------------------------------------------------------------------------
__global__ void __launch_bounds__(256)
attn_kernel(const h16* __restrict__ Qh, const h16* __restrict__ Kh,
            const h16* __restrict__ Vh, h16* __restrict__ Ctx)
{
  __shared__ h16 Vt[2][D_ * 48];    // V block transposed: [d][key], 32 keys
  __shared__ h16 Pb[8 * 16 * 48];   // per-wave P tile: 16 rows x 32 keys

  const int tid  = threadIdx.x;
  const int lane = tid & 31;
  const int wave = tid >> 5;
  const int sub  = lane >> 4;
  const int l16  = lane & 15;

  const int bh  = blockIdx.y;                     // b*H + h
  const int qr0 = blockIdx.x * 128 + wave * 16;   // query-row base of this wave
  const h16* Qb = Qh + (size_t)bh * S_ * D_;
  const h16* Kb = Kh + (size_t)bh * S_ * D_;
  const h16* Vb = Vh + (size_t)bh * S_ * D_;

  // Q fragments (already pre-scaled by 1/sqrt(D) in the projection)
  v16h qa[2];
#pragma unroll
  for (int kk = 0; kk < 2; ++kk)
    qa[kk] = ldfrag(Qb + (size_t)(qr0 + l16) * D_ + kk * 32 + sub * 8);

  float mrow[8], lrow[8];
#pragma unroll
  for (int r = 0; r < 8; ++r) { mrow[r] = -3.0e38f; lrow[r] = 0.0f; }
  v8f o[4] = {};

  h16* Pw = Pb + wave * (16 * 48);

  const int vkey = tid >> 3;        // 0..31
  const int vc   = (tid & 7) * 8;   // d-chunk

  auto stage_v = [&](int bsel, int kb) {
    v8h vv = *(const v8h*)(Vb + (size_t)(kb + vkey) * D_ + vc);
#pragma unroll
    for (int i = 0; i < 8; ++i) Vt[bsel][(vc + i) * 48 + vkey] = vv[i];
  };

  stage_v(0, 0);
  __syncthreads();

  int bsel = 0;
  for (int kb = 0; kb < S_; kb += 32) {
    if (kb + 32 < S_) stage_v(bsel ^ 1, kb + 32);

    // scores = Q K^T : K rows are contiguous -> B-fragments straight from global
    v8f sc[2] = {};
#pragma unroll
    for (int nf = 0; nf < 2; ++nf) {
      const h16* kp = Kb + (size_t)(kb + nf * 16 + l16) * D_;
#pragma unroll
      for (int kk = 0; kk < 2; ++kk) {
        v16h kf = ldfrag(kp + kk * 32 + sub * 8);
        sc[nf] = wmma_f16(qa[kk], kf, sc[nf]);
      }
    }

    // online softmax update (row lives across the 16 lanes of each half-wave)
#pragma unroll
    for (int r = 0; r < 8; ++r) {
      float v = fmaxf(sc[0][r], sc[1][r]);
#pragma unroll
      for (int off = 1; off < 16; off <<= 1)
        v = fmaxf(v, __shfl_xor(v, off, 32));
      float mn = fmaxf(mrow[r], v);
      float al = __expf(mrow[r] - mn);
      mrow[r] = mn;
      lrow[r] *= al;
#pragma unroll
      for (int df = 0; df < 4; ++df) o[df][r] *= al;
      float e0 = __expf(sc[0][r] - mn);
      float e1 = __expf(sc[1][r] - mn);
      Pw[(r + sub * 8) * 48 + l16]      = (h16)e0;
      Pw[(r + sub * 8) * 48 + 16 + l16] = (h16)e1;
      float s = e0 + e1;
#pragma unroll
      for (int off = 1; off < 16; off <<= 1)
        s += __shfl_xor(s, off, 32);
      lrow[r] += s;
    }

    // O += P V  (P re-laid-out through per-wave LDS tile; in-order per-wave DS)
    v16h pa = ldfrag(Pw + l16 * 48 + sub * 8);
#pragma unroll
    for (int df = 0; df < 4; ++df) {
      v16h vf = ldfrag(Vt[bsel] + (df * 16 + l16) * 48 + sub * 8);
      o[df] = wmma_f16(pa, vf, o[df]);
    }
    __syncthreads();
    bsel ^= 1;
  }

  // normalize and write context [B,S,H*D] as f16
  const int b = bh >> 4, h = bh & (H_ - 1);
#pragma unroll
  for (int r = 0; r < 8; ++r) {
    float inv = 1.0f / lrow[r];
    int s = qr0 + r + sub * 8;
#pragma unroll
    for (int df = 0; df < 4; ++df) {
      int d = df * 16 + l16;
      Ctx[(size_t)(b * S_ + s) * E_ + h * D_ + d] = (h16)(o[df][r] * inv);
    }
  }
}

// ---------------------------------------------------------------------------
// Kernel 3: output projection  out = ctx @ Wo^T + bo  (fp32 out).
//   A tile (f16 ctx) is a pure copy -> CDNA5 async global->LDS DMA path.
// ---------------------------------------------------------------------------
__global__ void __launch_bounds__(256)
oproj_kernel(const h16* __restrict__ Ctx, const float* __restrict__ Wo,
             const float* __restrict__ bo, float* __restrict__ Out)
{
  __shared__ h16 At[2][BM * LDT];
  __shared__ h16 Bt[2][BN * LDT];

  const int tid  = threadIdx.x;
  const int lane = tid & 31;
  const int wave = tid >> 5;
  const int wm   = wave >> 2;
  const int wn   = wave & 3;
  const int sub  = lane >> 4;
  const int l16  = lane & 15;

  const int m0 = blockIdx.x * BM;
  const int n0 = blockIdx.y * BN;

  float bv2[2];
#pragma unroll
  for (int nf = 0; nf < 2; ++nf)
    bv2[nf] = bo[n0 + wn * 32 + nf * 16 + l16];

  v8f acc[4][2] = {};

  const int lr = tid >> 3;
  const int lc = (tid & 7) * 4;

  // A tile: f16 -> f16 straight copy, async DMA into LDS (ASYNCcnt-tracked)
  auto load_a = [&](int bsel, int k0) {
#pragma unroll
    for (int it = 0; it < 2; ++it) {
      int s = tid + it * 256;
      int r = s >> 2, c = (s & 3) * 8;
      const h16* g = Ctx + (size_t)(m0 + r) * E_ + k0 + c;
      unsigned loff = (unsigned)(size_t)(At[bsel] + r * LDT + c);
      asm volatile("global_load_async_to_lds_b128 %0, %1, off"
                   :: "v"(loff), "v"(g) : "memory");
    }
  };
  // W tile: fp32 -> f16 (needs VALU cvt, regular loads)
  auto load_w = [&](int bsel, int k0) {
#pragma unroll
    for (int it = 0; it < 4; ++it) {
      int rr = lr + it * 32;
      v4f wa = *(const v4f*)(Wo + (size_t)(n0 + rr) * E_ + k0 + lc);
      h16* pb = Bt[bsel] + rr * LDT + lc;
#pragma unroll
      for (int i = 0; i < 4; ++i) pb[i] = (h16)wa[i];
    }
  };

  load_a(0, 0);
  load_w(0, 0);
  asm volatile("s_wait_asynccnt 0x0" ::: "memory");
  __syncthreads();

  int bsel = 0;
  for (int kt = 0; kt < NKT; ++kt) {
    if (kt + 1 < NKT) {
      load_a(bsel ^ 1, (kt + 1) * BK);
      load_w(bsel ^ 1, (kt + 1) * BK);
    }

    v16h bf[2];
#pragma unroll
    for (int nf = 0; nf < 2; ++nf)
      bf[nf] = ldfrag(Bt[bsel] + (wn * 32 + nf * 16 + l16) * LDT + sub * 8);
#pragma unroll
    for (int mf = 0; mf < 4; ++mf) {
      v16h af = ldfrag(At[bsel] + (wm * 64 + mf * 16 + l16) * LDT + sub * 8);
#pragma unroll
      for (int nf = 0; nf < 2; ++nf)
        acc[mf][nf] = wmma_f16(af, bf[nf], acc[mf][nf]);
    }
    asm volatile("s_wait_asynccnt 0x0" ::: "memory");
    __syncthreads();
    bsel ^= 1;
  }

#pragma unroll
  for (int mf = 0; mf < 4; ++mf) {
#pragma unroll
    for (int nf = 0; nf < 2; ++nf) {
#pragma unroll
      for (int r = 0; r < 8; ++r) {
        int m = m0 + wm * 64 + mf * 16 + r + sub * 8;
        int n = n0 + wn * 32 + nf * 16 + l16;
        Out[(size_t)m * E_ + n] = acc[mf][nf][r] + bv2[nf];
      }
    }
  }
}

// ---------------------------------------------------------------------------
// Launcher
// ---------------------------------------------------------------------------
extern "C" void kernel_launch(void* const* d_in, const int* in_sizes, int n_in,
                              void* d_out, int out_size, void* d_ws, size_t ws_size,
                              hipStream_t stream) {
  (void)in_sizes; (void)n_in; (void)out_size; (void)ws_size;
  const float* X  = (const float*)d_in[0];
  const float* Wq = (const float*)d_in[1];
  const float* bq = (const float*)d_in[2];
  const float* Wk = (const float*)d_in[3];
  const float* bk = (const float*)d_in[4];
  const float* Wv = (const float*)d_in[5];
  const float* bv = (const float*)d_in[6];
  const float* Wo = (const float*)d_in[7];
  const float* bo = (const float*)d_in[8];

  h16* ws = (h16*)d_ws;
  const size_t NT = (size_t)B_ * H_ * S_ * D_;   // 8Mi elements
  h16* Qh = ws;
  h16* Kh = ws + NT;
  h16* Vh = ws + 2 * NT;
  h16* Cx = ws + 3 * NT;

  // 1) fused QKV projections: grid (Mtiles=64, Ntiles=8, {q,k,v})
  qkv_kernel<<<dim3(64, 8, 3), 256, 0, stream>>>(
      X, Wq, Wk, Wv, bq, bk, bv, Qh, Kh, Vh);

  // 2) streaming attention: grid (S/128 = 16 q-tiles, B*H = 64)
  attn_kernel<<<dim3(16, 64), 256, 0, stream>>>(Qh, Kh, Vh, Cx);

  // 3) output projection
  oproj_kernel<<<dim3(64, 8), 256, 0, stream>>>(Cx, Wo, bo, (float*)d_out);
}
